// Contrastive_Loss_37933151159058
// MI455X (gfx1250) — matrix operand order repaired and checked by reading, not compiled, not run on hardware
//
#include <hip/hip_runtime.h>
#include <hip/hip_bf16.h>
#include <math.h>

typedef __attribute__((ext_vector_type(16))) __bf16 v16bf;
typedef __attribute__((ext_vector_type(8)))  float  v8f;

#define BS   8192
#define HID  2048
#define KT   (HID / 32)   // 64 k-tiles of K=32
#define RT   (BS / 16)    // 512 row-tiles of 16

// ---------------------------------------------------------------------------
// Kernel 1: row-normalize (fp32) and pack to bf16 in WMMA fragment layout.
// mode 0 = A-matrix layout (16x32, ISA 7.12.2), mode 1 = B-matrix layout.
// Packed tile = 512 bf16 (1KB): lane-major, 16 elements per lane, so a wave
// fragment load is one contiguous 32B read per lane (two b128s).
// ---------------------------------------------------------------------------
__global__ __launch_bounds__(256) void norm_pack_kernel(
    const float* __restrict__ Z, __bf16* __restrict__ out, int bmode)
{
    const int rb  = blockIdx.x;        // 16-row block, 0..511
    const int tid = threadIdx.x;
    const int r   = tid >> 4;          // local row 0..15
    const int c16 = tid & 15;          // column chunk 0..15 (128 cols each)
    const float* zrow = Z + (size_t)(rb * 16 + r) * HID;

    // sum of squares, 128 elements per thread
    float p = 0.f;
    #pragma unroll 4
    for (int c = c16 * 128; c < c16 * 128 + 128; ++c) {
        float v = zrow[c];
        p += v * v;
    }
    __shared__ float sp[256];
    __shared__ float rinv[16];
    sp[tid] = p;
    __syncthreads();
    if (c16 == 0) {
        float s = 0.f;
        #pragma unroll
        for (int i = 0; i < 16; ++i) s += sp[r * 16 + i];
        rinv[r] = 1.0f / fmaxf(sqrtf(s), 1e-8f);   // matches eps semantics
    }
    __syncthreads();
    const float inv = rinv[r];

    // pack 128 normalized values of this row into fragment layout
    for (int k = c16 * 128; k < c16 * 128 + 128; ++k) {
        __bf16 b = (__bf16)(zrow[k] * inv);
        int tk = k >> 5, kp = k & 31;
        int lane, j;
        if (bmode == 0) {                 // A 16x32: lanes 0-15 K{0..7,16..23}
            int hi16 = kp >> 4;           //          lanes16-31 K{8..15,24..31}
            int kk   = kp & 15;
            j    = (kk & 7) + 8 * hi16;
            lane = r + 16 * (kk >> 3);
        } else {                          // B 32x16: lanes 0-15 K{0..7,16..23}
            lane = r + 16 * ((kp >> 3) & 1); //       lanes16-31 K{8..15,24..31}
            j    = (kp & 7) + 8 * (kp >> 4);
        }
        out[((size_t)rb * KT + tk) * 512 + (size_t)lane * 16 + j] = b;
    }
}

// ---------------------------------------------------------------------------
// Kernel 2: fused GEMM (bf16 WMMA, f32 accum) + exp + row/col reductions.
// Block tile 128x256, 8 waves in 2x4 grid, wave tile 64x64 (4x4 fragments).
// Software-pipelined: k-step kt+1 fragments load while kt WMMAs execute.
// ---------------------------------------------------------------------------
__device__ __forceinline__ void load_frags(const __bf16* base, int t0, int kt,
                                           v16bf f[4])
{
    #pragma unroll
    for (int i = 0; i < 4; ++i)
        f[i] = *(const v16bf*)(base + ((size_t)(t0 + i) * KT + kt) * 512);
}

__device__ __forceinline__ void mma_step(v8f acc[4][4], const v16bf a[4],
                                         const v16bf b[4])
{
    #pragma unroll
    for (int mi = 0; mi < 4; ++mi)
        #pragma unroll
        for (int ni = 0; ni < 4; ++ni)
            acc[mi][ni] = __builtin_amdgcn_wmma_f32_16x16x32_bf16(
                false, a[mi], false, b[ni], (short)0, acc[mi][ni],
                false, false);
}

__global__ __launch_bounds__(256) void gemm_cl_kernel(
    const __bf16* __restrict__ Ap, const __bf16* __restrict__ Bp,
    float* __restrict__ rowsum, float* __restrict__ colsum,
    float* __restrict__ diag)
{
    const int bm   = blockIdx.x;       // 0..63 (M blocks of 128)
    const int bn   = blockIdx.y;       // 0..31 (N blocks of 256)
    const int tid  = threadIdx.x;
    const int wave = tid >> 5;
    const int lane = tid & 31;
    const int wm   = wave >> 2;        // 0..1
    const int wn   = wave & 3;         // 0..3
    const int m0t  = bm * 8 + wm * 4;  // first 16-row tile of this wave (M)
    const int n0t  = bn * 16 + wn * 4; // first 16-col tile of this wave (N)

    v8f acc[4][4] = {};

    const __bf16* a0 = Ap + (size_t)lane * 16;
    const __bf16* b0 = Bp + (size_t)lane * 16;

    v16bf aP[4], bP[4], aQ[4], bQ[4];
    load_frags(a0, m0t, 0, aP);
    load_frags(b0, n0t, 0, bP);

    for (int kt = 0; kt < KT; kt += 2) {
        // prefetch kt+1 while computing kt
        load_frags(a0, m0t, kt + 1, aQ);
        load_frags(b0, n0t, kt + 1, bQ);
        mma_step(acc, aP, bP);
        // prefetch kt+2 while computing kt+1 (clamped dummy reload at tail)
        const int k2 = (kt + 2 < KT) ? kt + 2 : kt;
        load_frags(a0, m0t, k2, aP);
        load_frags(b0, n0t, k2, bP);
        mma_step(acc, aQ, bQ);
    }

    // --- fused epilogue: e = exp(s); row/col partial sums; diagonal ---
    // C layout: lane<16 -> N=lane,    M = g      (VGPR g)
    //           lane>=16-> N=lane-16, M = 8 + g
    const int half = lane >> 4;
    const int l15  = lane & 15;
    const int gn0  = n0t * 16;

    float csum[4] = {0.f, 0.f, 0.f, 0.f};

    #pragma unroll
    for (int mi = 0; mi < 4; ++mi) {
        const int gm0 = (m0t + mi) * 16;
        #pragma unroll
        for (int g = 0; g < 8; ++g) {
            float rsum = 0.f;
            #pragma unroll
            for (int ni = 0; ni < 4; ++ni) {
                float s = acc[mi][ni][g];
                float e = __expf(s);
                csum[ni] += e;
                rsum     += e;
                // diagonal element: M == N inside a diagonal fragment
                if (gm0 == gn0 + 16 * ni && l15 == g + 8 * half)
                    diag[gm0 + l15] = s;
            }
            // row reduction across the 16 N-lanes of this half
            rsum += __shfl_xor(rsum, 1, 32);
            rsum += __shfl_xor(rsum, 2, 32);
            rsum += __shfl_xor(rsum, 4, 32);
            rsum += __shfl_xor(rsum, 8, 32);
            if (l15 == 0)
                atomicAdd(&rowsum[gm0 + g + 8 * half], rsum);
        }
    }
    // column sums: combine the two lane-halves (M 0-7 with M 8-15)
    #pragma unroll
    for (int ni = 0; ni < 4; ++ni) {
        float c = csum[ni] + __shfl_xor(csum[ni], 16, 32);
        if (half == 0)
            atomicAdd(&colsum[gn0 + 16 * ni + l15], c);
    }
}

// ---------------------------------------------------------------------------
// Kernel 3: out[i] = log(colsum_i - p) + log(rowsum_i - p) - 2*d,  p = exp(d)
// ---------------------------------------------------------------------------
__global__ __launch_bounds__(256) void finalize_kernel(
    const float* __restrict__ rowsum, const float* __restrict__ colsum,
    const float* __restrict__ diag, float* __restrict__ out)
{
    int i = blockIdx.x * 256 + threadIdx.x;
    if (i < BS) {
        float d = diag[i];
        float p = __expf(d);
        out[i] = logf(colsum[i] - p) + logf(rowsum[i] - p) - 2.0f * d;
    }
}

// ---------------------------------------------------------------------------
extern "C" void kernel_launch(void* const* d_in, const int* in_sizes, int n_in,
                              void* d_out, int out_size, void* d_ws, size_t ws_size,
                              hipStream_t stream)
{
    (void)in_sizes; (void)n_in; (void)out_size; (void)ws_size;
    const float* X = (const float*)d_in[0];
    const float* Y = (const float*)d_in[1];
    float* out = (float*)d_out;

    char* ws = (char*)d_ws;
    const size_t packBytes = (size_t)BS * HID * sizeof(__bf16); // 32 MB
    __bf16* Apack  = (__bf16*)(ws);
    __bf16* Bpack  = (__bf16*)(ws + packBytes);
    float*  rowsum = (float*)(ws + 2 * packBytes);
    float*  colsum = rowsum + BS;
    float*  diag   = colsum + BS;

    // zero the accumulators every call (graph replays don't re-poison)
    hipMemsetAsync(rowsum, 0, 2 * BS * sizeof(float), stream);

    norm_pack_kernel<<<RT, 256, 0, stream>>>(X, Apack, 0);
    norm_pack_kernel<<<RT, 256, 0, stream>>>(Y, Bpack, 1);

    gemm_cl_kernel<<<dim3(BS / 128, BS / 256), 256, 0, stream>>>(
        Apack, Bpack, rowsum, colsum, diag);

    finalize_kernel<<<BS / 256, 256, 0, stream>>>(rowsum, colsum, diag, out);
}
